// GATModel_47734266527996
// MI455X (gfx1250) — matrix operand order repaired and checked by reading, not compiled, not run on hardware
//
#include <hip/hip_runtime.h>

typedef __attribute__((ext_vector_type(16))) _Float16 v16h;
typedef __attribute__((ext_vector_type(8)))  _Float16 v8h;
typedef __attribute__((ext_vector_type(8)))  float    v8f;
typedef __attribute__((ext_vector_type(4)))  float    v4f;

#define GAT_N 50000
#define GAT_E 800000
#define GAT_H 4

// ---------------- utility fills ----------------
__global__ void fill_f32(float* __restrict__ p, float v, int n) {
    int i = blockIdx.x * blockDim.x + threadIdx.x;
    if (i < n) p[i] = v;
}
__global__ void fill_u32(unsigned* __restrict__ p, unsigned v, int n) {
    int i = blockIdx.x * blockDim.x + threadIdx.x;
    if (i < n) p[i] = v;
}

// ---------------- f32 -> f16 convert (activations, row-major kept) ----------------
__global__ void cvt_f16(const float* __restrict__ in, _Float16* __restrict__ out, int n) {
    int i = blockIdx.x * blockDim.x + threadIdx.x;
    if (i < n) out[i] = (_Float16)in[i];
}

// ---------------- weight convert + transpose: W[K][Fout] f32 -> Wt[Fout][K] f16 ----------------
__global__ void cvt_transpose_f16(const float* __restrict__ W, _Float16* __restrict__ Wt,
                                  int K, int Fout) {
    int i = blockIdx.x * blockDim.x + threadIdx.x;
    if (i >= K * Fout) return;
    int k = i / Fout, j = i % Fout;
    Wt[(size_t)j * K + k] = (_Float16)W[i];
}

// ---------------- layer-1 projection (K=2, VALU) ----------------
__global__ void gemm_k2(const float* __restrict__ feat, const float* __restrict__ W1,
                        float* __restrict__ z, int n_nodes, int Fout) {
    int i = blockIdx.x * blockDim.x + threadIdx.x;
    if (i >= n_nodes * Fout) return;
    int n = i / Fout, j = i % Fout;
    z[i] = fmaf(feat[2 * n], W1[j], feat[2 * n + 1] * W1[Fout + j]);
}

// ---------------- WMMA f16 GEMM: Z[N,Fout] = A[N,K] * B[K,Fout] ----------------
// Ah: f16 row-major [N,K].  Bt: f16 transposed [Fout][K].
// One wave computes a 16x32 output tile (2 adjacent N-tiles) reusing the A fragment.
__global__ void wmma_gemm_f16(const _Float16* __restrict__ Ah, const _Float16* __restrict__ Bt,
                              float* __restrict__ Z, int K, int Fout, int mtiles) {
    const int NT = 2;                               // N-tiles per wave
    int wavesPerBlock = blockDim.x >> 5;
    int wid = blockIdx.x * wavesPerBlock + (threadIdx.x >> 5);
    int ntg = (Fout >> 4) / NT;                     // tile-groups along N
    if (wid >= mtiles * ntg) return;                // whole-wave exit only (EXEC all-ones for WMMA)
    int mt   = wid / ntg;
    int ng   = wid % ntg;
    int lane = threadIdx.x & 31;
    int grp  = lane >> 4;                           // lane group 0/1
    int l15  = lane & 15;

    const _Float16* arow  = Ah + (size_t)(mt * 16 + l15) * K;
    const _Float16* bcol0 = Bt + (size_t)(ng * NT * 16 + l15) * K;
    const _Float16* bcol1 = bcol0 + (size_t)16 * K;

    v8f acc0 = {}, acc1 = {};
    for (int kb = 0; kb < K; kb += 32) {
        // A 16x32 f16 per-lane layout: contiguous run [grp*16, grp*16+16)
        v8h alo = *(const v8h*)(arow + kb + grp * 16);
        v8h ahi = *(const v8h*)(arow + kb + grp * 16 + 8);
        v16h a;
#pragma unroll
        for (int t = 0; t < 8; t++) { a[t] = alo[t]; a[8 + t] = ahi[t]; }
        // B 32x16 f16: lane's 16 halves contiguous in transposed weights (32B aligned)
        v16h b0 = *(const v16h*)(bcol0 + kb + grp * 16);
        v16h b1 = *(const v16h*)(bcol1 + kb + grp * 16);
        acc0 = __builtin_amdgcn_wmma_f32_16x16x32_f16(false, a, false, b0,
                                                      (short)0, acc0, false, false);
        acc1 = __builtin_amdgcn_wmma_f32_16x16x32_f16(false, a, false, b1,
                                                      (short)0, acc1, false, false);
    }
    // D layout: VGPR r -> row mt*16 + 8*grp + r, col base + l15
    float* zp = Z + (size_t)(mt * 16 + grp * 8) * Fout + ng * NT * 16 + l15;
#pragma unroll
    for (int r = 0; r < 8; r++) {
        zp[(size_t)r * Fout]      = acc0[r];
        zp[(size_t)r * Fout + 16] = acc1[r];
    }
}

// ---------------- per-node attention scores el/er ----------------
__global__ void attn_scores(const float* __restrict__ z, const float* __restrict__ al,
                            const float* __restrict__ ar, float* __restrict__ el,
                            float* __restrict__ er, int n_nodes, int Hh, int D) {
    int i = blockIdx.x * blockDim.x + threadIdx.x;
    if (i >= n_nodes * Hh) return;
    int n = i / Hh, h = i % Hh;
    const float* zp  = z + (size_t)n * Hh * D + (size_t)h * D;
    const float* alp = al + h * D;
    const float* arp = ar + h * D;
    float sl = 0.f, sr = 0.f;
    for (int d = 0; d < D; d++) {
        float zv = zp[d];
        sl = fmaf(zv, alp[d], sl);
        sr = fmaf(zv, arp[d], sr);
    }
    el[i] = sl;
    er[i] = sr;
}

// monotonic float<->uint key for atomic max on floats
__device__ __forceinline__ unsigned fkey(float f) {
    unsigned u = __float_as_uint(f);
    return (u & 0x80000000u) ? ~u : (u | 0x80000000u);
}
__device__ __forceinline__ float funkey(unsigned k) {
    unsigned u = (k & 0x80000000u) ? (k ^ 0x80000000u) : ~k;
    return __uint_as_float(u);
}

// ---------------- edge pass 1: segment max (L2 integer atomics) ----------------
__global__ void edge_max(const float* __restrict__ el, const float* __restrict__ er,
                         const int* __restrict__ src, const int* __restrict__ dst,
                         unsigned* __restrict__ mkey, int n_edges, int Hh) {
    int i = blockIdx.x * blockDim.x + threadIdx.x;
    if (i >= n_edges * Hh) return;
    int e = i / Hh, h = i % Hh;
    int s = src[e], d = dst[e];
    float v = el[s * Hh + h] + er[d * Hh + h];
    v = v > 0.f ? v : 0.2f * v;                 // leaky relu 0.2
    atomicMax(mkey + d * Hh + h, fkey(v));
}

// ---------------- edge pass 2: exp + segment sum ----------------
__global__ void edge_exp(const float* __restrict__ el, const float* __restrict__ er,
                         const int* __restrict__ src, const int* __restrict__ dst,
                         const unsigned* __restrict__ mkey, float* __restrict__ sden,
                         float* __restrict__ exv, int n_edges, int Hh) {
    int i = blockIdx.x * blockDim.x + threadIdx.x;
    if (i >= n_edges * Hh) return;
    int e = i / Hh, h = i % Hh;
    int s = src[e], d = dst[e];
    float v = el[s * Hh + h] + er[d * Hh + h];
    v = v > 0.f ? v : 0.2f * v;
    float m  = funkey(mkey[d * Hh + h]);
    float ex = __expf(v - m);
    exv[i] = ex;
    atomicAdd(sden + d * Hh + h, ex);
}

// ---------------- edge pass 3: gather z[src], weight by alpha, scatter-add to out[dst] ----------------
// one wave per edge; float4 (b128) gathers; f32 atomics resolve in L2 (arrays fit in 192MB L2)
__global__ void edge_aggregate(const float* __restrict__ z, const float* __restrict__ exv,
                               const float* __restrict__ sden, const int* __restrict__ src,
                               const int* __restrict__ dst, float* __restrict__ out,
                               int n_edges, int HD, int D, int Hh) {
    int wave = blockIdx.x * (blockDim.x >> 5) + (threadIdx.x >> 5);
    if (wave >= n_edges) return;
    int lane = threadIdx.x & 31;
    int s = src[wave], d = dst[wave];
    const float* zp = z + (size_t)s * HD;
    float*       op = out + (size_t)d * HD;
    for (int c = lane * 4; c < HD; c += 128) {
        v4f zv = *(const v4f*)(zp + c);
        int h = c / D;                          // 4 consecutive elems share a head (D % 4 == 0)
        float a = exv[(size_t)wave * Hh + h] / sden[d * Hh + h];
        atomicAdd(op + c + 0, zv[0] * a);
        atomicAdd(op + c + 1, zv[1] * a);
        atomicAdd(op + c + 2, zv[2] * a);
        atomicAdd(op + c + 3, zv[3] * a);
    }
}

// ---------------- bias + optional relu (in place) ----------------
__global__ void bias_act(float* __restrict__ out, const float* __restrict__ b,
                         int n, int HD, int relu) {
    int i = blockIdx.x * blockDim.x + threadIdx.x;
    if (i >= n) return;
    float v = out[i] + b[i % HD];
    if (relu) v = v > 0.f ? v : 0.f;
    out[i] = v;
}

// =====================================================================

static inline int cdiv(int a, int b) { return (a + b - 1) / b; }

extern "C" void kernel_launch(void* const* d_in, const int* in_sizes, int n_in,
                              void* d_out, int out_size, void* d_ws, size_t ws_size,
                              hipStream_t stream) {
    const float* feat = (const float*)d_in[0];
    const int*   src  = (const int*)d_in[1];
    const int*   dst  = (const int*)d_in[2];
    const float* W1   = (const float*)d_in[3];
    const float* al1  = (const float*)d_in[4];
    const float* ar1  = (const float*)d_in[5];
    const float* b1   = (const float*)d_in[6];
    const float* W2   = (const float*)d_in[7];
    const float* al2  = (const float*)d_in[8];
    const float* ar2  = (const float*)d_in[9];
    const float* b2   = (const float*)d_in[10];
    const float* W3   = (const float*)d_in[11];
    const float* al3  = (const float*)d_in[12];
    const float* ar3  = (const float*)d_in[13];
    const float* b3   = (const float*)d_in[14];
    float* out = (float*)d_out;

    const int N = GAT_N, E = GAT_E, H = GAT_H;

    // workspace carve-out
    size_t off = 0;
    char* ws = (char*)d_ws;
    auto carve = [&](size_t bytes) -> void* {
        void* p = ws + off;
        off += (bytes + 255) & ~(size_t)255;
        return p;
    };
    float*     zbuf = (float*)carve((size_t)N * 256 * sizeof(float));  // projections (max width 256)
    float*     hbuf = (float*)carve((size_t)N * 128 * sizeof(float));  // layer activations (f32)
    _Float16*  h16  = (_Float16*)carve((size_t)N * 128 * sizeof(_Float16)); // f16 activations for WMMA
    float*     el   = (float*)carve((size_t)N * H * sizeof(float));
    float*     er   = (float*)carve((size_t)N * H * sizeof(float));
    unsigned*  mkey = (unsigned*)carve((size_t)N * H * sizeof(unsigned));
    float*     sden = (float*)carve((size_t)N * H * sizeof(float));
    float*     exv  = (float*)carve((size_t)E * H * sizeof(float));
    _Float16*  W2t  = (_Float16*)carve((size_t)128 * 128 * sizeof(_Float16));
    _Float16*  W3t  = (_Float16*)carve((size_t)128 * 256 * sizeof(_Float16));

    const int TB = 256;
    const int mtiles = N / 16;                  // 3125 exact

    // weight conversion (cheap, done every call -> deterministic)
    cvt_transpose_f16<<<cdiv(128 * 128, TB), TB, 0, stream>>>(W2, W2t, 128, 128);
    cvt_transpose_f16<<<cdiv(128 * 256, TB), TB, 0, stream>>>(W3, W3t, 128, 256);

    // ---------------- layer 1: 2 -> 128 (H=4, D=32) ----------------
    gemm_k2<<<cdiv(N * 128, TB), TB, 0, stream>>>(feat, W1, zbuf, N, 128);
    attn_scores<<<cdiv(N * H, TB), TB, 0, stream>>>(zbuf, al1, ar1, el, er, N, H, 32);
    fill_u32<<<cdiv(N * H, TB), TB, 0, stream>>>(mkey, 0u, N * H);
    fill_f32<<<cdiv(N * H, TB), TB, 0, stream>>>(sden, 0.f, N * H);
    edge_max<<<cdiv(E * H, TB), TB, 0, stream>>>(el, er, src, dst, mkey, E, H);
    edge_exp<<<cdiv(E * H, TB), TB, 0, stream>>>(el, er, src, dst, mkey, sden, exv, E, H);
    fill_f32<<<cdiv(N * 128, TB), TB, 0, stream>>>(hbuf, 0.f, N * 128);
    edge_aggregate<<<cdiv(E, 8), TB, 0, stream>>>(zbuf, exv, sden, src, dst, hbuf, E, 128, 32, H);
    bias_act<<<cdiv(N * 128, TB), TB, 0, stream>>>(hbuf, b1, N * 128, 128, 1);

    // ---------------- layer 2: 128 -> 128 (H=4, D=32) ----------------
    cvt_f16<<<cdiv(N * 128, TB), TB, 0, stream>>>(hbuf, h16, N * 128);
    wmma_gemm_f16<<<cdiv(mtiles * (128 / 16 / 2), 8), TB, 0, stream>>>(h16, W2t, zbuf, 128, 128, mtiles);
    attn_scores<<<cdiv(N * H, TB), TB, 0, stream>>>(zbuf, al2, ar2, el, er, N, H, 32);
    fill_u32<<<cdiv(N * H, TB), TB, 0, stream>>>(mkey, 0u, N * H);
    fill_f32<<<cdiv(N * H, TB), TB, 0, stream>>>(sden, 0.f, N * H);
    edge_max<<<cdiv(E * H, TB), TB, 0, stream>>>(el, er, src, dst, mkey, E, H);
    edge_exp<<<cdiv(E * H, TB), TB, 0, stream>>>(el, er, src, dst, mkey, sden, exv, E, H);
    fill_f32<<<cdiv(N * 128, TB), TB, 0, stream>>>(hbuf, 0.f, N * 128);
    edge_aggregate<<<cdiv(E, 8), TB, 0, stream>>>(zbuf, exv, sden, src, dst, hbuf, E, 128, 32, H);
    bias_act<<<cdiv(N * 128, TB), TB, 0, stream>>>(hbuf, b2, N * 128, 128, 1);

    // ---------------- layer 3: 128 -> 256 (H=4, D=64), no relu ----------------
    cvt_f16<<<cdiv(N * 128, TB), TB, 0, stream>>>(hbuf, h16, N * 128);
    wmma_gemm_f16<<<cdiv(mtiles * (256 / 16 / 2), 8), TB, 0, stream>>>(h16, W3t, zbuf, 128, 256, mtiles);
    attn_scores<<<cdiv(N * H, TB), TB, 0, stream>>>(zbuf, al3, ar3, el, er, N, H, 64);
    fill_u32<<<cdiv(N * H, TB), TB, 0, stream>>>(mkey, 0u, N * H);
    fill_f32<<<cdiv(N * H, TB), TB, 0, stream>>>(sden, 0.f, N * H);
    edge_max<<<cdiv(E * H, TB), TB, 0, stream>>>(el, er, src, dst, mkey, E, H);
    edge_exp<<<cdiv(E * H, TB), TB, 0, stream>>>(el, er, src, dst, mkey, sden, exv, E, H);
    fill_f32<<<cdiv(N * 256, TB), TB, 0, stream>>>(out, 0.f, N * 256);
    edge_aggregate<<<cdiv(E, 8), TB, 0, stream>>>(zbuf, exv, sden, src, dst, out, E, 256, 64, H);
    bias_act<<<cdiv(N * 256, TB), TB, 0, stream>>>(out, b3, N * 256, 256, 0);
}
